// NRIEdgeLayer_4698694222005
// MI455X (gfx1250) — compile-verified, matrix operand order checked by reading, not run on hardware
//
#include <hip/hip_runtime.h>

typedef __attribute__((ext_vector_type(16))) _Float16 v16h;
typedef __attribute__((ext_vector_type(8)))  _Float16 v8h;
typedef __attribute__((ext_vector_type(8)))  float    v8f;

#define N_NODES 256
#define K_EDGE  8
#define D_EMB   128
#define H_DIM   64
#define K_MLP   9          // K_EDGE + 1 (allow_non_adjacent)
#define KH      (K_MLP * H_DIM)   // 576 per-node layer-0 activations
#define RS      584        // padded LDS row stride (halfs): 576+8 -> conflict-free b128

// ---------------- kernel 0: zero the (poisoned) output ----------------
__global__ void nri_zero_out(float* __restrict__ out) {
  int i = blockIdx.x * blockDim.x + threadIdx.x;
  if (i < N_NODES * H_DIM) out[i] = 0.0f;
}

// ---------------- kernel 1: layer-0 pre-GEMMs (tiny) ----------------
// h0sb[s][k][h] = node[s]·W0s[k][:,h] + b0[k][h]   (f16)
// h0d [d][k][h] = node[d]·W0d[k][:,h]              (f16)
// grid.x: 0..255 -> source rows (with bias), 256..511 -> dest rows
__global__ void nri_layer0(const float* __restrict__ node,
                           const float* __restrict__ w0s,
                           const float* __restrict__ w0d,
                           const float* __restrict__ b0,
                           _Float16* __restrict__ h0sb,
                           _Float16* __restrict__ h0d) {
  __shared__ float nrow[D_EMB];
  const int  bid   = blockIdx.x;
  const bool isSrc = bid < N_NODES;
  const int  n     = bid & (N_NODES - 1);
  const int  tid   = threadIdx.x;                 // 0..575
  if (tid < D_EMB) nrow[tid] = node[n * D_EMB + tid];
  __syncthreads();
  const int k = tid / H_DIM;
  const int h = tid - k * H_DIM;
  const float* w = isSrc ? w0s : w0d;
  float acc = isSrc ? b0[k * H_DIM + h] : 0.0f;
#pragma unroll 4
  for (int x = 0; x < D_EMB; ++x)
    acc = fmaf(nrow[x], w[(k * D_EMB + x) * H_DIM + h], acc);
  _Float16* o = isSrc ? h0sb : h0d;
  o[n * KH + k * H_DIM + h] = (_Float16)acc;
}

// ------- kernel 2: pack W1 (f32 [9][64][64]) into WMMA B-operand layout -------
// Layout [k][ntile t][kchunk c][lane][j] of f16, mirroring the documented 16-bit
// A layout: lanes 0-15 hold column y=t*16+lane with K=j (j<8) / j+8 (j>=8);
// lanes 16-31 hold the same columns with K offset +8.  One 32B load/lane fills
// a v16h operand in nri_main.
__global__ void nri_pack_w1(const float* __restrict__ w1,
                            _Float16* __restrict__ w1pk) {
  int idx  = blockIdx.x * blockDim.x + threadIdx.x;   // exactly 9*4*2*32*16
  int j    = idx & 15;
  int lane = (idx >> 4) & 31;
  int c    = (idx >> 9) & 1;
  int t    = (idx >> 10) & 3;
  int k    = idx >> 12;
  int y = t * 16 + (lane & 15);
  int x = c * 32 + ((lane >= 16) ? 8 : 0) + ((j < 8) ? j : j + 8);
  w1pk[idx] = (_Float16)w1[(k * H_DIM + x) * H_DIM + y];
}

// ---------------- kernel 3: main fused pairwise-MLP + message passing ----------
// Per wave: one 16-dest tile (WMMA M rows = dests), 8 sources, all 9 MLPs.
// A operand = relu(h0sb[s]+h0d[d]) built in packed f16; B = resident W1_k
// operands; C accumulates z1 with l1_bias folded into the initializer; the
// per-(s,d,k) edge weight is a per-row scale fused into the f32 accumulator.
__global__ void __launch_bounds__(128)
nri_main(const float* __restrict__ edge, const float* __restrict__ mask,
         const _Float16* __restrict__ g_h0sb, const _Float16* __restrict__ g_h0d,
         const _Float16* __restrict__ w1pk, const float* __restrict__ b1,
         float* __restrict__ out) {
  __shared__ __attribute__((aligned(16))) _Float16 h0d_lds[16 * RS];
  const int tid    = threadIdx.x;
  const int lane   = tid & 31;
  const int wid    = tid >> 5;            // 0..3
  const int ncol   = lane & 15;           // N column / A row within tile
  const int off8   = (lane >> 4) * 8;     // half-wave K/M offset
  const int d0     = blockIdx.x * 16;     // dest tile base
  const int s_base = blockIdx.y * 32 + wid * 8;

  // stage the 16-dest layer-0 tile (f16) into LDS via the CDNA5 direct
  // global->LDS async path (no VGPR round-trip, tracked by ASYNCcnt).
  // 16 rows x 72 16-byte chunks; both sides 16B aligned (1152B / 1168B rows).
  for (int c = tid; c < 16 * 72; c += 128) {
    int d = c / 72, r = c - d * 72;
    const _Float16* gp = g_h0d + (size_t)(d0 + d) * KH + r * 8;
    unsigned int ldsoff = (unsigned int)(uintptr_t)&h0d_lds[d * RS + r * 8];
    asm volatile("global_load_async_to_lds_b128 %0, %1, off"
                 :: "v"(ldsoff), "v"((unsigned long long)(uintptr_t)gp)
                 : "memory");
  }
  asm volatile("s_wait_asynccnt 0x0" ::: "memory");
  __syncthreads();

  float acc[4][8];
#pragma unroll
  for (int t = 0; t < 4; ++t)
#pragma unroll
    for (int v = 0; v < 8; ++v) acc[t][v] = 0.0f;

  for (int k = 0; k < K_MLP; ++k) {
    // W1_k as 8 register-resident B operands (4 N-tiles x 2 K-chunks)
    v16h bop[4][2];
#pragma unroll
    for (int t = 0; t < 4; ++t)
#pragma unroll
      for (int c = 0; c < 2; ++c)
        bop[t][c] = *(const v16h*)(w1pk + ((((k * 4 + t) * 2 + c) * 32 + lane) * 16));
    float b1v[4];
#pragma unroll
    for (int t = 0; t < 4; ++t) b1v[t] = b1[k * H_DIM + t * 16 + ncol];

    const bool kEdge = (k < K_EDGE);
    for (int si = 0; si < 8; ++si) {
      const int s = s_base + si;
      const _Float16* sbp = g_h0sb + (s * K_MLP + k) * H_DIM;
      const _Float16* ddp = &h0d_lds[ncol * RS + k * H_DIM];
      __builtin_prefetch(sbp + KH, 0, 0);   // next source's row -> global_prefetch_b8

      // ---- build A = relu(h0sb + h0d) in the 16-bit A-operand layout ----
      v16h a[2];
#pragma unroll
      for (int c = 0; c < 2; ++c) {
        const int hb = c * 32 + off8;
        v8h sa = *(const v8h*)(sbp + hb);
        v8h sb = *(const v8h*)(sbp + hb + 16);
        v8h da = *(const v8h*)(ddp + hb);
        v8h db = *(const v8h*)(ddp + hb + 16);
        v8h r0 = sa + da;
        v8h r1 = sb + db;
#pragma unroll
        for (int i = 0; i < 8; ++i) {
          r0[i] = r0[i] > (_Float16)0 ? r0[i] : (_Float16)0;
          r1[i] = r1[i] > (_Float16)0 ? r1[i] : (_Float16)0;
        }
        a[c] = __builtin_shufflevector(r0, r1, 0, 1, 2, 3, 4, 5, 6, 7,
                                       8, 9, 10, 11, 12, 13, 14, 15);
      }

      // ---- per-(s,d,k) MLP weights: edge embedding (k<8) or mask (k==8) ----
      float scl[8];
      if (kEdge) {
        const float* pe = edge + ((size_t)s * N_NODES + d0 + off8) * K_EDGE + k;
#pragma unroll
        for (int v = 0; v < 8; ++v) scl[v] = pe[v * K_EDGE];
      } else {
        const float* pm = mask + (size_t)s * N_NODES + d0 + off8;
#pragma unroll
        for (int v = 0; v < 8; ++v) scl[v] = pm[v];
      }

      // ---- layer 1: 16x64 @ 64x64 as 8 WMMAs, bias pre-seeded into C ----
#pragma unroll
      for (int t = 0; t < 4; ++t) {
        v8f ct;
#pragma unroll
        for (int v = 0; v < 8; ++v) ct[v] = b1v[t];
        ct = __builtin_amdgcn_wmma_f32_16x16x32_f16(false, a[0], false, bop[t][0],
                                                    (short)0, ct, false, false);
        ct = __builtin_amdgcn_wmma_f32_16x16x32_f16(false, a[1], false, bop[t][1],
                                                    (short)0, ct, false, false);
#pragma unroll
        for (int v = 0; v < 8; ++v) {
          float r = fmaxf(ct[v], 0.0f);               // relu(z1 + b1)
          acc[t][v] = fmaf(scl[v], r, acc[t][v]);     // weighted sum over s,k
        }
      }
    }
  }

  // ---- accumulate partials into out[d][y] (C layout: row = v + off8) ----
#pragma unroll
  for (int t = 0; t < 4; ++t)
#pragma unroll
    for (int v = 0; v < 8; ++v)
      atomicAdd(&out[(d0 + off8 + v) * H_DIM + t * 16 + ncol], acc[t][v]);
}

extern "C" void kernel_launch(void* const* d_in, const int* in_sizes, int n_in,
                              void* d_out, int out_size, void* d_ws, size_t ws_size,
                              hipStream_t stream) {
  (void)in_sizes; (void)n_in; (void)out_size; (void)ws_size;
  const float* edge = (const float*)d_in[0];   // [256,256,8]
  const float* node = (const float*)d_in[1];   // [256,128]
  const float* mask = (const float*)d_in[2];   // [256,256]
  const float* w0s  = (const float*)d_in[3];   // [9,128,64]
  const float* w0d  = (const float*)d_in[4];   // [9,128,64]
  const float* b0   = (const float*)d_in[5];   // [9,64]
  const float* w1   = (const float*)d_in[6];   // [9,64,64]
  const float* b1   = (const float*)d_in[7];   // [9,64]
  float* out = (float*)d_out;                  // [256,64]

  // workspace: h0sb (288KB) | h0d (288KB) | w1pk (72KB)  => 648KB total
  _Float16* h0sb = (_Float16*)d_ws;
  _Float16* h0d  = h0sb + N_NODES * KH;
  _Float16* w1pk = h0d + N_NODES * KH;

  nri_zero_out<<<dim3(64), dim3(256), 0, stream>>>(out);
  nri_layer0<<<dim3(2 * N_NODES), dim3(KH), 0, stream>>>(node, w0s, w0d, b0, h0sb, h0d);
  nri_pack_w1<<<dim3(144), dim3(256), 0, stream>>>(w1, w1pk);
  nri_main<<<dim3(16, 8), dim3(128), 0, stream>>>(edge, mask, h0sb, h0d, w1pk, b1, out);
}
